// IDMForwardSim_87402584473808
// MI455X (gfx1250) — compile-verified, matrix operand order checked by reading, not run on hardware
//
#include <hip/hip_runtime.h>

typedef __attribute__((ext_vector_type(16))) __bf16 v16bf;
typedef __attribute__((ext_vector_type(8)))  float  v8f;

#define ENC 50
#define TT 20
#define KP 64          // padded hidden (K) dim; rows 50/51/52 carry z0/z1/bias
#define NP 256         // padded gate columns: 4 gates x 64
#define RPW 16         // batch rows per wave
#define WPG 8          // waves per workgroup
#define RPG 128        // batch rows per workgroup

__device__ __forceinline__ unsigned short f2bf(float f) {
    unsigned int u = __float_as_uint(f);
    u += 0x7fffu + ((u >> 16) & 1u);          // round-to-nearest-even
    return (unsigned short)(u >> 16);
}
// hardware v_rcp_f32 (~1 ulp) instead of IEEE div sequence
__device__ __forceinline__ float fast_rcp(float x) { return __builtin_amdgcn_rcpf(x); }
__device__ __forceinline__ float sigmoidf_(float x) { return fast_rcp(1.0f + __expf(-x)); }
__device__ __forceinline__ float tanhf_(float x)    { return 2.0f * sigmoidf_(2.0f * x) - 1.0f; }

union ABFrag { v16bf v; uint4 q[2]; };

__global__ __launch_bounds__(256, 4) void idm_lstm_kernel(
    const float* __restrict__ s,        // (B,T,7)
    const float* __restrict__ z,        // (B,2)
    const float* __restrict__ desired_v,
    const float* __restrict__ desired_tgap,
    const float* __restrict__ min_jamx,
    const float* __restrict__ max_act,
    const float* __restrict__ min_act,
    const float* __restrict__ h0,       // (B,50)
    const float* __restrict__ c0,       // (B,50)
    const float* __restrict__ W_lstm,   // (2,200)
    const float* __restrict__ U_lstm,   // (50,200)
    const float* __restrict__ b_lstm,   // (200)
    const float* __restrict__ W_att,    // (50,50)
    const float* __restrict__ b_att,    // (50)
    const float* __restrict__ W_neu,    // (50,1)
    const float* __restrict__ b_neu,    // (1)
    float* __restrict__ out,            // [act (B*T)] ++ [att (B*T)]
    int B)
{
    // U stored column-major per padded output column n: U_s[n*64 + k], bf16 (32KB).
    // Rows 50,51,52 hold W_lstm[0], W_lstm[1], b_lstm so the WMMA computes xz+b too.
    __shared__ __attribute__((aligned(16))) unsigned short U_s[NP * KP];
    // per-wave h tile, row-major 16x64 bf16 (2KB per wave, 16KB total);
    // cols 50,51 = z0,z1; col 52 = 1.0 (written once, never overwritten)
    __shared__ __attribute__((aligned(16))) unsigned short h_s[WPG * RPW * KP];
    __shared__ float wc_s[KP];                         // W_att @ W_neu (padded)
    __shared__ float bcomb_s;
    __shared__ float att_s[WPG][RPW];

    const int tid  = threadIdx.x;
    const int lane = tid & 31;
    const int wv   = tid >> 5;
    const int half = lane >> 4;          // which 16-lane half
    const int nn   = lane & 15;
    const int wave_base = blockIdx.x * RPG + wv * RPW;

    // ---------------- one-time preprocessing ----------------
    {
        const int n = tid;               // one padded column per thread (0..255)
        const int gate = n >> 6, cl = n & 63;
        const bool valid = cl < ENC;
        const int col = gate * ENC + cl; // original column in (.,200)
        for (int k = 0; k < KP; ++k) {
            float v = 0.0f;
            if (valid) {
                if (k < ENC)           v = U_lstm[k * 200 + col];
                else if (k == ENC)     v = W_lstm[col];          // z0 row
                else if (k == ENC + 1) v = W_lstm[200 + col];    // z1 row
                else if (k == ENC + 2) v = b_lstm[col];          // bias row
            }
            U_s[n * KP + k] = f2bf(v);
        }
        if (tid < KP) {
            float acc = 0.0f;
            if (tid < ENC)
                for (int k2 = 0; k2 < ENC; ++k2) acc += W_att[tid * ENC + k2] * W_neu[k2];
            wc_s[tid] = acc;
        }
        if (tid == 0) {
            float acc = 0.0f;
            for (int k2 = 0; k2 < ENC; ++k2) acc += b_att[k2] * W_neu[k2];
            bcomb_s = acc + b_neu[0];
        }
        // h0 (+ z, + 1.0) -> per-wave LDS tile
        for (int it = 0; it < 32; ++it) {
            int idx = it * 32 + lane;                 // 0..1023
            int row = idx >> 6, j = idx & 63;
            float v = 0.0f;
            if (j < ENC)           v = h0[(size_t)(wave_base + row) * ENC + j];
            else if (j == ENC)     v = z[(size_t)(wave_base + row) * 2 + 0];
            else if (j == ENC + 1) v = z[(size_t)(wave_base + row) * 2 + 1];
            else if (j == ENC + 2) v = 1.0f;
            h_s[(wv * RPW + row) * KP + j] = f2bf(v);
        }
    }
    __syncthreads();

    // ---------------- persistent per-lane state ----------------
    // cell state in C/D fragment layout: cst[slice][r] is (m = r+8*half, j = 16*slice+nn)
    float cst[4][8];
    #pragma unroll
    for (int sl = 0; sl < 4; ++sl) {
        int j = 16 * sl + nn;
        #pragma unroll
        for (int r = 0; r < 8; ++r) {
            int m = r + 8 * half;
            cst[sl][r] = (j < ENC) ? c0[(size_t)(wave_base + m) * ENC + j] : 0.0f;
        }
    }
    // per-batch IDM parameters (lanes 0..15 each own batch row wave_base+lane)
    float p_dvinv = 0.f, p_tgap = 0.f, p_jamx = 0.f, p_maxa = 0.f, p_coef = 0.f;
    if (lane < 16) {
        int b = wave_base + lane;
        p_dvinv = fast_rcp(desired_v[b]);
        p_tgap  = desired_tgap[b];
        p_jamx  = min_jamx[b];
        p_maxa  = max_act[b];
        p_coef  = 0.5f * fast_rcp(sqrtf(p_maxa * min_act[b]));
    }

    const unsigned short* hrow = h_s + (wv * RPW + nn) * KP;   // A-frag source row (m = lane&15)
    unsigned short* hst = h_s + wv * RPW * KP;
    const int coff = half * 8;                                 // 16B chunk select per half

    for (int t = 0; t < TT; ++t) {
        // Stop LICM from hoisting the (loop-invariant) B-fragment LDS loads into
        // 256 live VGPRs and spilling them to scratch: force per-iteration reload.
        asm volatile("" ::: "memory");

        // A fragments for h_t (16x64 bf16): contiguous 16B chunks per half
        ABFrag a0, a1;
        a0.q[0] = *reinterpret_cast<const uint4*>(hrow + coff);
        a0.q[1] = *reinterpret_cast<const uint4*>(hrow + 16 + coff);
        a1.q[0] = *reinterpret_cast<const uint4*>(hrow + 32 + coff);
        a1.q[1] = *reinterpret_cast<const uint4*>(hrow + 48 + coff);

        float att_part[8];
        #pragma unroll
        for (int r = 0; r < 8; ++r) att_part[r] = 0.0f;

        #pragma unroll
        for (int sl = 0; sl < 4; ++sl) {           // hidden-unit slice: j = 16*sl + nn
            v8f acc[4];                            // i, f, g, o gate tiles
            #pragma unroll
            for (int g = 0; g < 4; ++g) {
                const int tile = g * 4 + sl;
                const int ncol = tile * 16 + nn;
                const unsigned short* bcol = U_s + ncol * KP;     // column-major B source
                ABFrag b0, b1;
                b0.q[0] = *reinterpret_cast<const uint4*>(bcol + coff);
                b0.q[1] = *reinterpret_cast<const uint4*>(bcol + 16 + coff);
                b1.q[0] = *reinterpret_cast<const uint4*>(bcol + 32 + coff);
                b1.q[1] = *reinterpret_cast<const uint4*>(bcol + 48 + coff);
                v8f cz = {0.f, 0.f, 0.f, 0.f, 0.f, 0.f, 0.f, 0.f};
                acc[g] = __builtin_amdgcn_wmma_f32_16x16x32_bf16(
                    false, a0.v, false, b0.v, (short)0, cz, false, false);
                acc[g] = __builtin_amdgcn_wmma_f32_16x16x32_bf16(
                    false, a1.v, false, b1.v, (short)0, acc[g], false, false);
            }
            const float wcv = wc_s[16 * sl + nn];
            #pragma unroll
            for (int r = 0; r < 8; ++r) {
                float iv = sigmoidf_(acc[0][r]);
                float fv = sigmoidf_(acc[1][r]);
                float gv = tanhf_(acc[2][r]);
                float ov = sigmoidf_(acc[3][r]);
                float cn = fv * cst[sl][r] + iv * gv;
                cst[sl][r] = cn;
                float hn = ov * tanhf_(cn);
                att_part[r] += hn * wcv;
                int m = r + 8 * half;
                // keep cols 50..63 (z / 1.0 / zero pad) intact: slice 3 stores only j=48,49
                if (sl != 3 || nn < 2)
                    hst[m * KP + 16 * sl + nn] = f2bf(hn);
            }
        }

        // attention: reduce over hidden dim (16 lanes per half hold distinct j's)
        #pragma unroll
        for (int r = 0; r < 8; ++r) {
            float v = att_part[r];
            v += __shfl_xor(v, 1, 32);
            v += __shfl_xor(v, 2, 32);
            v += __shfl_xor(v, 4, 32);
            v += __shfl_xor(v, 8, 32);
            att_part[r] = v;
        }
        #pragma unroll
        for (int r = 0; r < 8; ++r)
            if (nn == r) att_s[wv][r + 8 * half] = att_part[r];   // one writer per m

        // output phase: lanes 0..15 own one batch row each
        if (lane < 16) {
            int b = wave_base + lane;
            float att = sigmoidf_(5.0f * (att_s[wv][lane] + bcomb_s));
            const float* sp = s + ((size_t)b * TT + t) * 7;
            float vel = sp[0];
            float dvl = sp[2], dxl = sp[3];
            float dvm = sp[5], dxm = sp[6];
            float common = p_jamx + p_tgap * vel;
            float v4 = vel * p_dvinv; v4 *= v4; v4 *= v4;
            float gl = common + vel * dvl * p_coef;
            float gm = common + vel * dvm * p_coef;
            float rl = gl * fast_rcp(dxl), rm = gm * fast_rcp(dxm);
            float fl = p_maxa * (1.0f - v4 - rl * rl);
            float fm = p_maxa * (1.0f - v4 - rm * rm);
            fl = fminf(fmaxf(fl, -3.5f), 3.5f);
            fm = fminf(fmaxf(fm, -3.5f), 3.5f);
            float act = att * fl + (1.0f - att) * fm;
            out[(size_t)b * TT + t] = act;
            out[(size_t)B * TT + (size_t)b * TT + t] = att;
        }
    }
}

extern "C" void kernel_launch(void* const* d_in, const int* in_sizes, int n_in,
                              void* d_out, int out_size, void* d_ws, size_t ws_size,
                              hipStream_t stream) {
    (void)n_in; (void)d_ws; (void)ws_size; (void)out_size;
    const float* s       = (const float*)d_in[0];
    const float* z       = (const float*)d_in[1];
    const float* dv      = (const float*)d_in[2];
    const float* tg      = (const float*)d_in[3];
    const float* jx      = (const float*)d_in[4];
    const float* mx      = (const float*)d_in[5];
    const float* mn      = (const float*)d_in[6];
    const float* h0      = (const float*)d_in[7];
    const float* c0      = (const float*)d_in[8];
    const float* W_lstm  = (const float*)d_in[9];
    const float* U_lstm  = (const float*)d_in[10];
    const float* b_lstm  = (const float*)d_in[11];
    const float* W_att   = (const float*)d_in[12];
    const float* b_att   = (const float*)d_in[13];
    const float* W_neu   = (const float*)d_in[14];
    const float* b_neu   = (const float*)d_in[15];

    const int B = in_sizes[2];                 // desired_v has B elements
    dim3 grid((B + RPG - 1) / RPG), block(256);
    idm_lstm_kernel<<<grid, block, 0, stream>>>(
        s, z, dv, tg, jx, mx, mn, h0, c0,
        W_lstm, U_lstm, b_lstm, W_att, b_att, W_neu, b_neu,
        (float*)d_out, B);
}
